// AugmentedAttentionHead_87471303950972
// MI455X (gfx1250) — matrix operand order repaired and checked by reading, not compiled, not run on hardware
//
#include <hip/hip_runtime.h>
#include <hip/hip_bf16.h>

#define B_   64
#define T_   577
#define E_   768
#define H_   64
#define GW_  24
#define TPAD 608      // 19*32 padded key rows
#define T16  38       // TPAD/16
#define T32  19       // TPAD/32
#define QT   37       // ceil(577/16) query tiles

typedef __attribute__((ext_vector_type(16))) __bf16 v16bf;
typedef __attribute__((ext_vector_type(8)))  float  v8f;

union FragU {
  v16bf v;
  unsigned int   ui[8];
  unsigned short us[16];
};

__device__ inline v8f v8f_zero() { v8f z; for (int i = 0; i < 8; ++i) z[i] = 0.f; return z; }

// f32 -> bf16 via the hardware convert path
__device__ inline unsigned short f2bfu(float f) {
  __bf16 h = (__bf16)f;
  return __builtin_bit_cast(unsigned short, h);
}
__device__ inline unsigned int f2bf_pk(float lo, float hi) {
  return (unsigned int)f2bfu(lo) | ((unsigned int)f2bfu(hi) << 16);
}

__device__ inline v8f wmma_bf16(v16bf a, v16bf b, v8f c) {
  return __builtin_amdgcn_wmma_f32_16x16x32_bf16(false, a, false, b, (short)0, c, false, false);
}

// butterfly reductions within 16-lane halves (rows of C layout stay per-half)
__device__ inline float rsum16(float v) {
  v += __shfl_xor(v, 1, 32); v += __shfl_xor(v, 2, 32);
  v += __shfl_xor(v, 4, 32); v += __shfl_xor(v, 8, 32);
  return v;
}
__device__ inline float rmax16(float v) {
  v = fmaxf(v, __shfl_xor(v, 1, 32)); v = fmaxf(v, __shfl_xor(v, 2, 32));
  v = fmaxf(v, __shfl_xor(v, 4, 32)); v = fmaxf(v, __shfl_xor(v, 8, 32));
  return v;
}

__device__ inline void ld_frag32(FragU& f, const uint4* base, size_t idx) {
  uint4 a = base[idx], b = base[idx + 1];
  f.ui[0] = a.x; f.ui[1] = a.y; f.ui[2] = a.z; f.ui[3] = a.w;
  f.ui[4] = b.x; f.ui[5] = b.y; f.ui[6] = b.z; f.ui[7] = b.w;
}

// ---------------------------------------------------------------------------
// Stage 0: relayout [w_q|w_k|w_v] (f32, (768,64) each) into bf16 WMMA
// B-fragment layout: WB[kc(24)][nc(12)][lane(32)][16]
// ---------------------------------------------------------------------------
__global__ void relayout_w_kernel(const float* __restrict__ wq,
                                  const float* __restrict__ wk,
                                  const float* __restrict__ wv,
                                  unsigned short* __restrict__ WB) {
  int gid = blockIdx.x * blockDim.x + threadIdx.x;   // 24*12*32 = 9216
  if (gid >= 24 * 12 * 32) return;
  int lane = gid & 31;
  int frag = gid >> 5;
  int nc = frag % 12, kc = frag / 12;
  int n = nc * 16 + (lane & 15);
  int hi = lane >> 4;
  const float* w; int col;
  if (n < 64)       { w = wq; col = n; }
  else if (n < 128) { w = wk; col = n - 64; }
  else              { w = wv; col = n - 128; }
  unsigned short* dst = WB + (((size_t)kc * 12 + nc) * 32 + lane) * 16;
#pragma unroll
  for (int e = 0; e < 16; ++e) {
    int k = 32 * kc + ((e < 8) ? (8 * hi + e) : (16 + 8 * hi + (e - 8)));
    dst[e] = f2bfu(w[(size_t)k * 64 + col]);
  }
}

// ---------------------------------------------------------------------------
// Stage 1: fused QKV GEMM (WMMA bf16) + LayerNorm(q,k) + alpha/sigma gates.
// 256-thread blocks: 8 waves = 8 row tiles share one double-buffered 12KB
// WB slice in LDS (b-independent), per-thread staging is only 3 uint4.
// ---------------------------------------------------------------------------
__global__ void __launch_bounds__(256)
qkv_ln_kernel(const float* __restrict__ x, const unsigned short* __restrict__ WB,
              const float* __restrict__ q_gamma, const float* __restrict__ q_beta,
              const float* __restrict__ k_gamma, const float* __restrict__ k_beta,
              const float* __restrict__ w_sigma, const float* __restrict__ b_sigma,
              const float* __restrict__ w_alpha, const float* __restrict__ b_alpha,
              unsigned short* __restrict__ qbf, unsigned short* __restrict__ kB,
              unsigned short* __restrict__ vB, float* __restrict__ alpha_o,
              float* __restrict__ i2sx_o, float* __restrict__ i2sy_o) {
  __shared__ uint4 WLds[2][768];                    // 2 x 12KB
  const int tid  = threadIdx.x;
  const int wave = tid >> 5;
  const int lane = tid & 31;
  const int tile = blockIdx.x * 8 + wave;            // 64*38 tiles total
  const int b    = tile / T16;
  const int t16  = tile - b * T16;
  const int row0 = t16 * 16;
  const int n    = lane & 15;
  const int hi   = lane >> 4;

  const int  tA = row0 + n;                          // A-matrix row for this lane
  const bool rowvalid = (tA < T_);
  const float* xrow = x + ((size_t)b * T_ + (rowvalid ? tA : 0)) * E_;

  v8f acc[12];
#pragma unroll
  for (int i = 0; i < 12; ++i) acc[i] = v8f_zero();

  const uint4* wb4 = reinterpret_cast<const uint4*>(WB);

  // preload kc = 0 slice (768 uint4 = 12KB)
#pragma unroll
  for (int j = 0; j < 3; ++j) WLds[0][tid + 256 * j] = wb4[tid + 256 * j];
  __syncthreads();

  for (int kc = 0; kc < 24; ++kc) {
    // kick off global loads of next WB slice (consumed after compute)
    uint4 g[3];
    if (kc + 1 < 24) {
#pragma unroll
      for (int j = 0; j < 3; ++j)
        g[j] = wb4[(size_t)(kc + 1) * 768 + tid + 256 * j];
    }
    // build A fragment (x f32 -> bf16, hardware cvt)
    FragU a;
    if (rowvalid) {
      const float* p0 = xrow + 32 * kc + 8 * hi;
      const float* p1 = xrow + 32 * kc + 16 + 8 * hi;
#pragma unroll
      for (int j = 0; j < 4; ++j) a.ui[j] = f2bf_pk(p0[2 * j], p0[2 * j + 1]);
#pragma unroll
      for (int j = 0; j < 4; ++j) a.ui[4 + j] = f2bf_pk(p1[2 * j], p1[2 * j + 1]);
    } else {
#pragma unroll
      for (int e = 0; e < 8; ++e) a.ui[e] = 0u;
    }
    // consume current slice from LDS
    const uint4* lbuf = &WLds[kc & 1][0];
#pragma unroll
    for (int nc = 0; nc < 12; ++nc) {
      FragU bf;
      ld_frag32(bf, lbuf, (size_t)(nc * 64 + lane * 2));
      acc[nc] = wmma_bf16(a.v, bf.v, acc[nc]);
    }
    if (kc + 1 < 24) {
      __syncthreads();                               // everyone done reading buf^1
#pragma unroll
      for (int j = 0; j < 3; ++j) WLds[(kc + 1) & 1][tid + 256 * j] = g[j];
      __syncthreads();                               // stores visible
    }
  }

  // per-lane column params (C layout: col = 16*nc + n)
  float gq[4], bq[4], gk[4], bk[4], wa[4], ws0[4], ws1[4];
#pragma unroll
  for (int nc = 0; nc < 4; ++nc) {
    int h = 16 * nc + n;
    gq[nc] = q_gamma[h]; bq[nc] = q_beta[h];
    gk[nc] = k_gamma[h]; bk[nc] = k_beta[h];
    wa[nc] = w_alpha[h];
    ws0[nc] = w_sigma[h * 2 + 0]; ws1[nc] = w_sigma[h * 2 + 1];
  }
  const float b_al = b_alpha[0], b_s0 = b_sigma[0], b_s1 = b_sigma[1];

#pragma unroll
  for (int r = 0; r < 8; ++r) {
    // LayerNorm q (in place)
    float s = 0.f, s2 = 0.f;
#pragma unroll
    for (int nc = 0; nc < 4; ++nc) { float vv = acc[nc][r]; s += vv; s2 += vv * vv; }
    s = rsum16(s); s2 = rsum16(s2);
    float mu = s * 0.015625f;
    float rstd = rsqrtf(s2 * 0.015625f - mu * mu + 1e-5f);
#pragma unroll
    for (int nc = 0; nc < 4; ++nc)
      acc[nc][r] = (acc[nc][r] - mu) * rstd * gq[nc] + bq[nc];
    // LayerNorm k (in place)
    s = 0.f; s2 = 0.f;
#pragma unroll
    for (int nc = 0; nc < 4; ++nc) { float vv = acc[4 + nc][r]; s += vv; s2 += vv * vv; }
    s = rsum16(s); s2 = rsum16(s2);
    mu = s * 0.015625f;
    rstd = rsqrtf(s2 * 0.015625f - mu * mu + 1e-5f);
#pragma unroll
    for (int nc = 0; nc < 4; ++nc)
      acc[4 + nc][r] = (acc[4 + nc][r] - mu) * rstd * gk[nc] + bk[nc];
    // gates from normalized q
    float pa = 0.f, p0 = 0.f, p1 = 0.f;
#pragma unroll
    for (int nc = 0; nc < 4; ++nc) {
      float qv = acc[nc][r];
      pa += qv * wa[nc]; p0 += qv * ws0[nc]; p1 += qv * ws1[nc];
    }
    pa = rsum16(pa); p0 = rsum16(p0); p1 = rsum16(p1);
    if (n == 0) {
      int t = row0 + r + 8 * hi;
      float alin = pa + b_al;
      float al = (alin > 20.f) ? alin : log1pf(__expf(alin));
      float sx = 1.f / (1.f + __expf(-(p0 + b_s0)));
      float sy = 1.f / (1.f + __expf(-(p1 + b_s1)));
      size_t idx = (size_t)b * TPAD + t;
      alpha_o[idx] = al;
      i2sx_o[idx] = 1.f / (2.f * sx * sx);
      i2sy_o[idx] = 1.f / (2.f * sy * sy);
    }
  }

  // store q row-major bf16
  unsigned short* qd = qbf + ((size_t)b * TPAD + row0) * 64;
#pragma unroll
  for (int nc = 0; nc < 4; ++nc)
#pragma unroll
    for (int r = 0; r < 8; ++r) {
      int M = r + 8 * hi;
      qd[(size_t)M * 64 + 16 * nc + n] = f2bfu(acc[nc][r]);
    }
  // store k into B-fragment layout: kB[b][t16][c][lane'=16*h2+M][e], hl=h&31: h2=(hl>>3)&1, e=(hl&7)+8*(hl>>4)
#pragma unroll
  for (int nc = 0; nc < 4; ++nc)
#pragma unroll
    for (int r = 0; r < 8; ++r) {
      int M = r + 8 * hi;
      int h = 16 * nc + n;
      int c = h >> 5, h2 = (h >> 3) & 1, e = (h & 7) + 8 * ((h >> 4) & 1);
      size_t idx = ((((size_t)b * T16 + t16) * 2 + c) * 32 + 16 * h2 + M) * 16 + e;
      kB[idx] = f2bfu(acc[4 + nc][r]);
    }
  // store v into B-fragment layout: vB[b][t32][hc][lane'=16*(M>>3)+n][e=(M&7)+8*(t16&1)]
#pragma unroll
  for (int nc = 0; nc < 4; ++nc)
#pragma unroll
    for (int r = 0; r < 8; ++r) {
      int M = r + 8 * hi;
      size_t idx = ((((size_t)b * T32 + (t16 >> 1)) * 4 + nc) * 32 + 16 * (M >> 3) + n) * 16 + (M & 7) + 8 * (t16 & 1);
      vB[idx] = f2bfu(acc[8 + nc][r]);
    }
}

// ---------------------------------------------------------------------------
// Stage 2: flash attention with Gaussian positional bias.
// Grid (5, 64): blockIdx.y = batch, 8 waves = 8 query tiles sharing one
// batch, so the 8KB of k/v fragments per key tile is staged via LDS once
// per block (double-buffered, 1 uint4 per thread per array).
// ---------------------------------------------------------------------------
__global__ void __launch_bounds__(256)
attn_kernel(const unsigned short* __restrict__ qbf,
            const unsigned short* __restrict__ kBp,
            const unsigned short* __restrict__ vBp,
            const float* __restrict__ alpha_a, const float* __restrict__ i2sx_a,
            const float* __restrict__ i2sy_a, float* __restrict__ out) {
  __shared__ uint4 KLds[2][256];                    // 2 x 4KB
  __shared__ uint4 VLds[2][256];                    // 2 x 4KB
  __shared__ unsigned short Pscr[8][16 * 32];       // 8 x 1KB
  const int tid  = threadIdx.x;
  const int wave = tid >> 5;
  const int lane = tid & 31;
  const int b    = blockIdx.y;
  const int qtr  = blockIdx.x * 8 + wave;            // 0..39
  const int qt   = (qtr < QT) ? qtr : (QT - 1);      // clamp; dup waves skip store
  const int qr0  = qt * 16;
  const int n    = lane & 15;
  const int hi   = lane >> 4;

  // q A-fragments for both K-chunks of H=64
  FragU qa[2];
  {
    const unsigned short* qrow = qbf + ((size_t)b * TPAD + qr0 + n) * 64;
#pragma unroll
    for (int c = 0; c < 2; ++c) {
      const uint4* p0 = reinterpret_cast<const uint4*>(qrow + 32 * c + 8 * hi);
      const uint4* p1 = reinterpret_cast<const uint4*>(qrow + 32 * c + 16 + 8 * hi);
      uint4 u0 = *p0, u1 = *p1;
      qa[c].ui[0] = u0.x; qa[c].ui[1] = u0.y; qa[c].ui[2] = u0.z; qa[c].ui[3] = u0.w;
      qa[c].ui[4] = u1.x; qa[c].ui[5] = u1.y; qa[c].ui[6] = u1.z; qa[c].ui[7] = u1.w;
    }
  }

  // per-row (C layout rows M = r + 8*hi) gate params and grid coords
  float alP[8], ixP[8], iyP[8], qxP[8], qyP[8], mrow[8], lrow[8];
  int rowt[8]; bool qv[8];
#pragma unroll
  for (int r = 0; r < 8; ++r) {
    int t = qr0 + r + 8 * hi;
    rowt[r] = t;
    qv[r] = (t >= 1) && (t < T_);
    size_t idx = (size_t)b * TPAD + t;
    alP[r] = alpha_a[idx]; ixP[r] = i2sx_a[idx]; iyP[r] = i2sy_a[idx];
    int qi = (t >= 1) ? (t - 1) : 0;
    qyP[r] = (float)(qi / GW_);
    qxP[r] = (float)(qi % GW_);
    mrow[r] = -3.0e38f; lrow[r] = 0.f;
  }
  v8f o[4];
#pragma unroll
  for (int hc = 0; hc < 4; ++hc) o[hc] = v8f_zero();

  const uint4* kb4 = reinterpret_cast<const uint4*>(kBp);
  const uint4* vb4 = reinterpret_cast<const uint4*>(vBp);
  unsigned short* ps = &Pscr[wave][0];

  // preload key tile 0: kB region (4KB) + vB region (4KB), 1 uint4 each
  KLds[0][tid] = kb4[((size_t)(b * T16) * 2) * 64 + tid];
  VLds[0][tid] = vb4[((size_t)(b * T32) * 4) * 64 + tid];
  __syncthreads();

  for (int kt = 0; kt < T32; ++kt) {
    // global loads for next key tile (consumed after compute)
    uint4 gk, gv;
    if (kt + 1 < T32) {
      gk = kb4[((size_t)(b * T16 + 2 * (kt + 1)) * 2) * 64 + tid];
      gv = vb4[((size_t)(b * T32 + (kt + 1)) * 4) * 64 + tid];
    }

    // sim = q x k^T from LDS-staged fragments
    const uint4* kl = &KLds[kt & 1][0];
    FragU kf[4];
#pragma unroll
    for (int c = 0; c < 2; ++c) {
      ld_frag32(kf[2 * c + 0], kl, (size_t)((0 * 2 + c) * 64 + lane * 2));
      ld_frag32(kf[2 * c + 1], kl, (size_t)((1 * 2 + c) * 64 + lane * 2));
    }
    v8f s0 = v8f_zero(), s1 = v8f_zero();
    s0 = wmma_bf16(qa[0].v, kf[0].v, s0);
    s1 = wmma_bf16(qa[0].v, kf[1].v, s1);
    s0 = wmma_bf16(qa[1].v, kf[2].v, s0);
    s1 = wmma_bf16(qa[1].v, kf[3].v, s1);

    // Gaussian bias + 1/sqrt(H) scale + key masking
    auto gauss = [&](v8f& s, int chunk) {
      int key = kt * 32 + 16 * chunk + n;
      bool kvalid = key < T_;
      int kj = (key >= 1) ? (key - 1) : 0;
      float ky = (float)(kj / GW_), kx = (float)(kj % GW_);
      bool kg = kvalid && (key >= 1);
#pragma unroll
      for (int r = 0; r < 8; ++r) {
        float z = s[r];
        if (kg && qv[r]) {
          float dx = qxP[r] - kx, dy = qyP[r] - ky;
          z += alP[r] * __expf(-dx * dx * ixP[r] - dy * dy * iyP[r]);
        }
        z *= 0.125f;
        s[r] = kvalid ? z : -1.0e30f;
      }
    };
    gauss(s0, 0);
    gauss(s1, 1);

    // online softmax
#pragma unroll
    for (int r = 0; r < 8; ++r) {
      float rm = rmax16(fmaxf(s0[r], s1[r]));
      float mnew = fmaxf(mrow[r], rm);
      float corr = __expf(mrow[r] - mnew);
      s0[r] = __expf(s0[r] - mnew);
      s1[r] = __expf(s1[r] - mnew);
      float rs = rsum16(s0[r] + s1[r]);
      lrow[r] = lrow[r] * corr + rs;
      mrow[r] = mnew;
      o[0][r] *= corr; o[1][r] *= corr; o[2][r] *= corr; o[3][r] *= corr;
    }

    // P (16x32 f32, C layout) -> LDS bf16 -> A fragment (per-wave scratch)
#pragma unroll
    for (int r = 0; r < 8; ++r) {
      int M = r + 8 * hi;
      ps[M * 32 + n]      = f2bfu(s0[r]);
      ps[M * 32 + 16 + n] = f2bfu(s1[r]);
    }
    asm volatile("s_wait_dscnt 0" ::: "memory");
    FragU pf;
#pragma unroll
    for (int e = 0; e < 8; ++e) pf.us[e] = ps[n * 32 + 8 * hi + e];
#pragma unroll
    for (int e = 0; e < 8; ++e) pf.us[8 + e] = ps[n * 32 + 16 + 8 * hi + e];
    asm volatile("s_wait_dscnt 0" ::: "memory");

    // O += P x V from LDS-staged fragments
    const uint4* vl = &VLds[kt & 1][0];
#pragma unroll
    for (int hc = 0; hc < 4; ++hc) {
      FragU vf;
      ld_frag32(vf, vl, (size_t)(hc * 64 + lane * 2));
      o[hc] = wmma_bf16(pf.v, vf.v, o[hc]);
    }

    // publish next tile into the other LDS buffer
    if (kt + 1 < T32) {
      __syncthreads();                               // all waves done with buf^1
      KLds[(kt + 1) & 1][tid] = gk;
      VLds[(kt + 1) & 1][tid] = gv;
      __syncthreads();                               // stores visible
    }
  }

  // epilogue: divide by row sums, store fp32 output (skip duplicate waves)
  if (qtr < QT) {
#pragma unroll
    for (int r = 0; r < 8; ++r) {
      int t = rowt[r];
      if (t < T_) {
        float inv = 1.f / lrow[r];
        size_t ob = ((size_t)b * T_ + t) * 64;
#pragma unroll
        for (int hc = 0; hc < 4; ++hc)
          out[ob + 16 * hc + n] = o[hc][r] * inv;
      }
    }
  }
}

// ---------------------------------------------------------------------------
// Workspace layout (bytes, 256-aligned): total ~15.7 MB
// ---------------------------------------------------------------------------
#define OFF_WB  0u
#define OFF_QBF 294912u
#define OFF_KB  (OFF_QBF + 4980736u)
#define OFF_VB  (OFF_KB + 4980736u)
#define OFF_AL  (OFF_VB + 4980736u)
#define OFF_SX  (OFF_AL + 155648u)
#define OFF_SY  (OFF_SX + 155648u)

extern "C" void kernel_launch(void* const* d_in, const int* in_sizes, int n_in,
                              void* d_out, int out_size, void* d_ws, size_t ws_size,
                              hipStream_t stream) {
  const float* x       = (const float*)d_in[0];
  const float* w_q     = (const float*)d_in[1];
  const float* w_k     = (const float*)d_in[2];
  const float* w_v     = (const float*)d_in[3];
  const float* q_gamma = (const float*)d_in[4];
  const float* q_beta  = (const float*)d_in[5];
  const float* k_gamma = (const float*)d_in[6];
  const float* k_beta  = (const float*)d_in[7];
  const float* w_sigma = (const float*)d_in[8];
  const float* b_sigma = (const float*)d_in[9];
  const float* w_alpha = (const float*)d_in[10];
  const float* b_alpha = (const float*)d_in[11];
  float* out = (float*)d_out;

  char* ws = (char*)d_ws;
  unsigned short* WB  = (unsigned short*)(ws + OFF_WB);
  unsigned short* qbf = (unsigned short*)(ws + OFF_QBF);
  unsigned short* kB  = (unsigned short*)(ws + OFF_KB);
  unsigned short* vB  = (unsigned short*)(ws + OFF_VB);
  float* alpha_o = (float*)(ws + OFF_AL);
  float* i2sx_o  = (float*)(ws + OFF_SX);
  float* i2sy_o  = (float*)(ws + OFF_SY);

  relayout_w_kernel<<<36, 256, 0, stream>>>(w_q, w_k, w_v, WB);
  qkv_ln_kernel<<<(B_ * T16) / 8, 256, 0, stream>>>(
      x, WB, q_gamma, q_beta, k_gamma, k_beta, w_sigma, b_sigma, w_alpha,
      b_alpha, qbf, kB, vB, alpha_o, i2sx_o, i2sy_o);
  attn_kernel<<<dim3(5, 64), 256, 0, stream>>>(qbf, kB, vB, alpha_o, i2sx_o,
                                               i2sy_o, out);
}
